// AttetionLayer_52673478918296
// MI455X (gfx1250) — matrix operand order repaired and checked by reading, not compiled, not run on hardware
//
#include <hip/hip_runtime.h>
#include <hip/hip_bf16.h>
#include <math.h>

#define DD 2048
#define TT 16384
#define TM 128
#define TN 128
#define TK 64
#define LDSS 72   // padded bf16 row stride (144B) -> conflict-free 16-lane row access

typedef __bf16 bf16x4  __attribute__((ext_vector_type(4)));
typedef __bf16 bf16x8  __attribute__((ext_vector_type(8)));
typedef __bf16 bf16x16 __attribute__((ext_vector_type(16)));
typedef float  f32x8   __attribute__((ext_vector_type(8)));

// ---------------- K1: qb[d] = dot(W1_w[d,:], query) + W1_b[d] + W2_b[d] ----------------
__global__ __launch_bounds__(256) void qproj_kernel(const float* __restrict__ query,
                                                    const float* __restrict__ W1w,
                                                    const float* __restrict__ W1b,
                                                    const float* __restrict__ W2b,
                                                    float* __restrict__ qb) {
    const int wave = threadIdx.x >> 5;
    const int lane = threadIdx.x & 31;
    const int row  = blockIdx.x * 8 + wave;
    const float* wrow = W1w + (size_t)row * DD;
    float s = 0.f;
    for (int k = lane; k < DD; k += 32) s += wrow[k] * query[k];
#pragma unroll
    for (int off = 16; off >= 1; off >>= 1) s += __shfl_xor(s, off, 32);
    if (lane == 0) qb[row] = s + W1b[row] + W2b[row];
}

// ---------------- K2: fused GEMM + tanh + exp, per-tile column partial sums ----------------
// grid = (DD/TN, TT/TM), block = 256 (8 waves; wave (wm,wn) owns 64Mx32N = 4x2 16x16 tiles)
__global__ __launch_bounds__(256) void attn_gemm_kernel(const float* __restrict__ values,
                                                        const float* __restrict__ W2w,
                                                        const float* __restrict__ qb,
                                                        const float* __restrict__ vvec,
                                                        float* __restrict__ elog,    // [T,D] = d_out
                                                        float* __restrict__ partial) // [T/128, D]
{
    __shared__ __bf16 ldsA[TM * LDSS];
    __shared__ __bf16 ldsB[TN * LDSS];
    __shared__ float  colpart[TN];

    const int tid   = threadIdx.x;
    const int lane  = tid & 31;
    const int wave  = tid >> 5;
    const int wm    = wave >> 2;   // 0..1 -> 64-row half
    const int wn    = wave & 3;    // 0..3 -> 32-col slice
    const int n0    = blockIdx.x * TN;
    const int m0    = blockIdx.y * TM;
    const int lhalf = lane >> 4;   // 0/1
    const int lcol  = lane & 15;

    const f32x8 zero8 = {0.f,0.f,0.f,0.f,0.f,0.f,0.f,0.f};
    f32x8 acc[4][2];
#pragma unroll
    for (int mi = 0; mi < 4; ++mi)
#pragma unroll
        for (int ni = 0; ni < 2; ++ni) acc[mi][ni] = zero8;

    for (int kc = 0; kc < DD; kc += TK) {
        if (kc) __syncthreads();           // previous compute done before overwriting LDS
        // cooperative fill: 128 rows x 16 float4 of A and B, converting f32 -> bf16
#pragma unroll
        for (int i = 0; i < 8; ++i) {
            const int idx = tid + i * 256;     // 0..2047
            const int rr  = idx >> 4;          // 0..127
            const int c4  = idx & 15;          // 0..15
            const float* pa = values + (size_t)(m0 + rr) * DD + kc + c4 * 4;
            const float* pb = W2w    + (size_t)(n0 + rr) * DD + kc + c4 * 4;
            const float4 va = *(const float4*)pa;
            const float4 vb = *(const float4*)pb;
            bf16x4 ba = { (__bf16)va.x, (__bf16)va.y, (__bf16)va.z, (__bf16)va.w };
            bf16x4 bb = { (__bf16)vb.x, (__bf16)vb.y, (__bf16)vb.z, (__bf16)vb.w };
            *(bf16x4*)(&ldsA[rr * LDSS + c4 * 4]) = ba;
            *(bf16x4*)(&ldsB[rr * LDSS + c4 * 4]) = bb;
            if (kc + TK < DD) {                // global_prefetch_b8 next K-chunk
                __builtin_prefetch(pa + TK, 0, 1);
                __builtin_prefetch(pb + TK, 0, 1);
            }
        }
        __syncthreads();

#pragma unroll
        for (int ks = 0; ks < 2; ++ks) {       // two 16x16x32 K-steps per chunk
            bf16x16 afrag[4];
            bf16x16 bfrag[2];
            // A (16x32 bf16): lane<16 -> K {0..7,16..23}; lane>=16 -> K {8..15,24..31}
#pragma unroll
            for (int mi = 0; mi < 4; ++mi) {
                const int ra = wm * 64 + mi * 16 + lcol;
                const bf16x8 lo = *(const bf16x8*)(&ldsA[ra * LDSS + ks * 32 + lhalf * 8]);
                const bf16x8 hi = *(const bf16x8*)(&ldsA[ra * LDSS + ks * 32 + lhalf * 8 + 16]);
                afrag[mi] = __builtin_shufflevector(lo, hi, 0,1,2,3,4,5,6,7,8,9,10,11,12,13,14,15);
            }
            // B (32x16 bf16): lane<16 -> K {0..7,16..23} of col lcol; lane>=16 -> K {8..15,24..31}
#pragma unroll
            for (int ni = 0; ni < 2; ++ni) {
                const int rb = wn * 32 + ni * 16 + lcol;
                const bf16x8 lo = *(const bf16x8*)(&ldsB[rb * LDSS + ks * 32 + lhalf * 8]);
                const bf16x8 hi = *(const bf16x8*)(&ldsB[rb * LDSS + ks * 32 + lhalf * 8 + 16]);
                bfrag[ni] = __builtin_shufflevector(lo, hi, 0,1,2,3,4,5,6,7,8,9,10,11,12,13,14,15);
            }
#pragma unroll
            for (int mi = 0; mi < 4; ++mi)
#pragma unroll
                for (int ni = 0; ni < 2; ++ni)
                    acc[mi][ni] = __builtin_amdgcn_wmma_f32_16x16x32_bf16(
                        false, afrag[mi], false, bfrag[ni],
                        (short)0, acc[mi][ni], false, false);
        }
    }

    __syncthreads();
    if (tid < TN) colpart[tid] = 0.f;
    __syncthreads();

    // epilogue: e = exp(tanh(acc + qb[col]) * v[col]); write e, accumulate column sums
    float csum[2] = {0.f, 0.f};
#pragma unroll
    for (int ni = 0; ni < 2; ++ni) {
        const int col = n0 + wn * 32 + ni * 16 + lcol;
        const float q  = qb[col];
        const float vv = vvec[col];
#pragma unroll
        for (int mi = 0; mi < 4; ++mi) {
            const int mbase = m0 + wm * 64 + mi * 16 + lhalf * 8;  // C layout: VGPR j -> row lhalf*8+j
#pragma unroll
            for (int j = 0; j < 8; ++j) {
                const float x = tanhf(acc[mi][ni][j] + q);
                const float e = __expf(x * vv);
                elog[(size_t)(mbase + j) * DD + col] = e;
                csum[ni] += e;
            }
        }
    }
#pragma unroll
    for (int ni = 0; ni < 2; ++ni) {
        csum[ni] += __shfl_xor(csum[ni], 16, 32);   // combine the two row-halves (same column)
        if (lhalf == 0)                              // exactly 2 adds/col (wm=0,1): commutative
            atomicAdd(&colpart[wn * 32 + ni * 16 + lcol], csum[ni]);
    }
    __syncthreads();
    if (tid < TN) partial[(size_t)blockIdx.y * DD + n0 + tid] = colpart[tid];
}

// ---------------- K3: deterministic column reduction -> 1/sum ----------------
__global__ __launch_bounds__(256) void colsum_kernel(const float* __restrict__ partial,
                                                     float* __restrict__ invsum) {
    const int d = blockIdx.x * 256 + threadIdx.x;
    float s = 0.f;
    for (int b = 0; b < TT / TM; ++b) s += partial[(size_t)b * DD + d];
    invsum[d] = 1.0f / s;
}

// ---------------- K4: out = e * values * invsum[col], in-place on d_out ----------------
__global__ __launch_bounds__(256) void finalize_kernel(const float* __restrict__ values,
                                                       const float* __restrict__ invsum,
                                                       float* __restrict__ out) {
    const size_t i = (size_t)blockIdx.x * 256 + threadIdx.x;     // float4 index
    const float4 e  = ((const float4*)out)[i];
    const float4 va = ((const float4*)values)[i];
    const float4 iv = ((const float4*)invsum)[i & (DD / 4 - 1)];
    float4 r;
    r.x = e.x * va.x * iv.x;
    r.y = e.y * va.y * iv.y;
    r.z = e.z * va.z * iv.z;
    r.w = e.w * va.w * iv.w;
    ((float4*)out)[i] = r;
}

extern "C" void kernel_launch(void* const* d_in, const int* in_sizes, int n_in,
                              void* d_out, int out_size, void* d_ws, size_t ws_size,
                              hipStream_t stream) {
    const float* query  = (const float*)d_in[0];
    const float* values = (const float*)d_in[1];
    const float* v      = (const float*)d_in[2];
    const float* W1w    = (const float*)d_in[3];
    const float* W1b    = (const float*)d_in[4];
    const float* W2w    = (const float*)d_in[5];
    const float* W2b    = (const float*)d_in[6];
    float* out = (float*)d_out;

    float* ws      = (float*)d_ws;
    float* qb      = ws;            // D floats
    float* invsum  = ws + DD;       // D floats
    float* partial = ws + 2 * DD;   // (T/128)*D floats (~1 MB)

    qproj_kernel<<<DD / 8, 256, 0, stream>>>(query, W1w, W1b, W2b, qb);

    dim3 g(DD / TN, TT / TM);
    attn_gemm_kernel<<<g, 256, 0, stream>>>(values, W2w, qb, v, out, partial);

    colsum_kernel<<<DD / 256, 256, 0, stream>>>(partial, invsum);

    finalize_kernel<<<(size_t)TT * DD / 4 / 256, 256, 0, stream>>>(values, invsum, out);
}
